// AU_GAZE_Affect7_LSTM_MLP_9088150799070
// MI455X (gfx1250) — matrix operand order repaired and checked by reading, not compiled, not run on hardware
//
#include <hip/hip_runtime.h>

// ---------------------------------------------------------------------------
// AU/GAZE/EXPR/VA 3-layer BiLSTM (H=8) + MLP head, for gfx1250 (CDNA5, wave32)
// Per-wave 16-batch tile; gates via v_wmma_f32_16x16x32_f16; f16 activations.
// D templated -> branchless loads; one ds_swizzle per gate row; fast rcp.
// ---------------------------------------------------------------------------

typedef _Float16 v16h __attribute__((ext_vector_type(16)));
typedef _Float16 v8h  __attribute__((ext_vector_type(8)));
typedef float    v8f  __attribute__((ext_vector_type(8)));

#define TSEQ  512
#define BATCH 2048
#define CCH   50
#define HID   8
#define WPB   4      // waves per block

__device__ __forceinline__ float fast_rcp(float x)  { return __builtin_amdgcn_rcpf(x); }
__device__ __forceinline__ float sigmf(float x)     { return fast_rcp(1.0f + __expf(-x)); }
__device__ __forceinline__ float tanh_fast(float x) { return 1.0f - 2.0f * fast_rcp(__expf(2.0f * x) + 1.0f); }
// lane(j) <-> lane(j^8) exchange: ds_swizzle group-of-32, xor=8, and=0x1f
__device__ __forceinline__ float swz_xor8(float v) {
    return __int_as_float(__builtin_amdgcn_ds_swizzle(__float_as_int(v), 0x201F));
}

// ---- raw x gather (fp32, strided by TSEQ) for K-range [KB, KB+16) split by lane half.
template<int D, int KB>
__device__ __forceinline__ void loadRaw8(float* dst, const float* __restrict__ xb, int hi) {
    if constexpr (KB < D) {
#pragma unroll
        for (int e = 0; e < 8; ++e) {
            const int k = KB + hi * 8 + e;
            const int ks = (KB + 15 < D) ? k : (k < D ? k : 0);   // clamp, mask later
            dst[e] = xb[(size_t)ks * TSEQ];
        }
    }
}
// ---- convert raw -> A-operand elems [off, off+8) with compile-time masking.
template<int D, int KB>
__device__ __forceinline__ void cvt8(v16h& A, int off, const float* src, int hi) {
#pragma unroll
    for (int e = 0; e < 8; ++e) {
        if constexpr (KB + 15 < D) {
            A[off + e] = (_Float16)src[e];
        } else if constexpr (KB >= D) {
            A[off + e] = (_Float16)0.0f;
        } else {
            const int k = KB + hi * 8 + e;
            A[off + e] = (k < D) ? (_Float16)src[e] : (_Float16)0.0f;
        }
    }
}

// One (branch, layer) pass, both directions (blockIdx.y). Each wave owns a
// 16-batch tile and runs the full sequential time loop with c/h in registers.
template<int D, bool FROM_X, bool WRITE_SEQ>
__global__ __launch_bounds__(WPB * 32)
void lstm_layer_kernel(const float* __restrict__ x, int chan_off,
                       const _Float16* __restrict__ in_seq,              // [T,B,16] f16 (when !FROM_X)
                       const float* __restrict__ wih_f, const float* __restrict__ whh_f,
                       const float* __restrict__ bih_f, const float* __restrict__ bhh_f,
                       const float* __restrict__ wih_b, const float* __restrict__ whh_b,
                       const float* __restrict__ bih_b, const float* __restrict__ bhh_b,
                       _Float16* __restrict__ out_seq,                   // [T,B,16] f16 (WRITE_SEQ)
                       float* __restrict__ out_last)                    // hcat+branch_off (!WRITE_SEQ)
{
    __shared__ __align__(16) _Float16 lds_h[WPB * 16 * HID];

    const int lane = threadIdx.x & 31;
    const int wave = threadIdx.x >> 5;
    const int dir  = blockIdx.y;                 // 0 = fwd, 1 = bwd
    const int b0   = (blockIdx.x * WPB + wave) * 16;

    const float* wih = dir ? wih_b : wih_f;
    const float* whh = dir ? whh_b : whh_f;
    const float* bih = dir ? bih_b : bih_f;
    const float* bhh = dir ? bhh_b : bhh_f;
    const int dirOff = dir ? HID : 0;

    const int nloc = lane & 15;     // N column within tile / A-matrix row m
    const int hi   = lane >> 4;     // lane-half (A/B K-split)
    const int j    = lane & 7;
    _Float16* myLds = lds_h + wave * 16 * HID;

    // ---- Weights into WMMA B-layout registers (clamped loads, cndmask masking).
    // 16-bit B 32x16: elems 0..7: K = hi*8+e ; elems 8..15: K = 16+hi*8+(e-8); N = lane&15.
    v16h whhB[2], wihB0[2], wihB1[2];
    float biasv[2];
#pragma unroll
    for (int nt = 0; nt < 2; ++nt) {
        const int g = nt * 16 + nloc;            // gate rows: 0-7 i, 8-15 f, 16-23 g, 24-31 o
        biasv[nt] = bih[g] + bhh[g];
#pragma unroll
        for (int e = 0; e < 16; ++e) {
            const int k  = (e < 8) ? (hi * 8 + e) : (16 + hi * 8 + (e - 8));
            const int k1 = 32 + k;
            const float wh = whh[g * HID + (k  < HID ? k : 0)];
            const float w0 = wih[g * D   + (k  < D   ? k  : 0)];
            whhB[nt][e]  = (k < HID) ? (_Float16)wh : (_Float16)0.0f;
            wihB0[nt][e] = (k < D)   ? (_Float16)w0 : (_Float16)0.0f;
            if constexpr (D > 32) {
                const float w1 = wih[g * D + (k1 < D ? k1 : 0)];
                wihB1[nt][e] = (k1 < D) ? (_Float16)w1 : (_Float16)0.0f;
            } else {
                wihB1[nt][e] = (_Float16)0.0f;
            }
        }
    }

    // ---- Recurrent state: h in A-layout; c-state lives on hi-8 lanes.
    v16h hA;
#pragma unroll
    for (int e = 0; e < 16; ++e) hA[e] = (_Float16)0.0f;
    float cst[8];
#pragma unroll
    for (int r = 0; r < 8; ++r) cst[r] = 0.0f;

    const int b_m = b0 + nloc;                   // batch row this lane supplies for A

    // ---- Raw next-step input (loads issued early, converted at top of iter).
    float raw[D > 32 ? 24 : 8];
    v8h rawRow;
    const float* xb0 = FROM_X ? (x + ((size_t)b_m * CCH + chan_off) * TSEQ) : nullptr;

    auto loadRaw = [&](int step) {
        const int t = dir ? (TSEQ - 1 - step) : step;
        if constexpr (FROM_X) {
            const float* xb = xb0 + t;
            loadRaw8<D, 0>(raw, xb, hi);
            loadRaw8<D, 16>(raw + 8, xb, hi);
            if constexpr (D > 32) loadRaw8<D, 32>(raw + 16, xb, hi);
        } else {
            rawRow = *(const v8h*)(in_seq + ((size_t)t * BATCH + b_m) * 16 + hi * 8);
        }
    };
    loadRaw(0);

    for (int step = 0; step < TSEQ; ++step) {
        const int t = dir ? (TSEQ - 1 - step) : step;

        // ---- Convert staged raw input into A operand(s). 16-bit A 16x32 layout:
        // elems 0..7: K = hi*8+e ; elems 8..15: K = 16+hi*8+e ; M = lane&15.
        v16h xA0, xA1;
        if constexpr (FROM_X) {
            cvt8<D, 0>(xA0, 0, raw, hi);
            cvt8<D, 16>(xA0, 8, raw + 8, hi);
            if constexpr (D > 32) {
                cvt8<D, 32>(xA1, 0, raw + 16, hi);
                cvt8<D, 48>(xA1, 8, raw + 16, hi);   // KB>=D -> zeros
            }
        } else {
#pragma unroll
            for (int e = 0; e < 8; ++e) { xA0[e] = rawRow[e]; xA0[8 + e] = (_Float16)0.0f; }
        }

        // ---- Gates: D = x*Wih^T + h*Whh^T + (bih+bhh), two 16-wide N-tiles.
        v8f dm0, dm1;
        {
            v8f acc;
#pragma unroll
            for (int e = 0; e < 8; ++e) acc[e] = biasv[0];
            acc = __builtin_amdgcn_wmma_f32_16x16x32_f16(false, xA0, false, wihB0[0], (short)0, acc, false, false);
            if constexpr (D > 32)
                acc = __builtin_amdgcn_wmma_f32_16x16x32_f16(false, xA1, false, wihB1[0], (short)0, acc, false, false);
            dm0 = __builtin_amdgcn_wmma_f32_16x16x32_f16(false, hA, false, whhB[0], (short)0, acc, false, false);
        }
        {
            v8f acc;
#pragma unroll
            for (int e = 0; e < 8; ++e) acc[e] = biasv[1];
            acc = __builtin_amdgcn_wmma_f32_16x16x32_f16(false, xA0, false, wihB0[1], (short)0, acc, false, false);
            if constexpr (D > 32)
                acc = __builtin_amdgcn_wmma_f32_16x16x32_f16(false, xA1, false, wihB1[1], (short)0, acc, false, false);
            dm1 = __builtin_amdgcn_wmma_f32_16x16x32_f16(false, hA, false, whhB[1], (short)0, acc, false, false);
        }

        // ---- Issue next step's input loads now (converted next iteration).
        if (step + 1 < TSEQ) loadRaw(step + 1);

        // ---- Activations. C/D layout: vgpr r, lane l -> (M = r + (l>>4)*8, N = l&15).
        // N-tile0 cols: 0-7 = i, 8-15 = f ; N-tile1: 0-7 = g, 8-15 = o.
        // lo-8 lanes compute p = sig(i)*tanh(g); hi-8 lanes own c: c = sig(f)*c + p,
        // h = sig(o)*tanh(c). One ds_swizzle (xor 8) per row.
        _Float16 hreg[8];
#pragma unroll
        for (int r = 0; r < 8; ++r) {
            const float d0v = dm0[r], d1v = dm1[r];
            const float e0 = sigmf(d0v);            // lo: sig(i) ; hi: sig(f)
            const float p  = e0 * tanh_fast(d1v);   // lo lanes: sig(i)*tanh(g)
            const float q  = swz_xor8(p);           // hi lanes receive partner's p
            const float c  = e0 * cst[r] + q;       // valid on hi-8 lanes
            cst[r] = c;
            hreg[r] = (_Float16)(sigmf(d1v) * tanh_fast(c));  // hi: sig(o)*tanh(c)
        }
        if (lane & 8) {                              // lanes 8-15 (M=r), 24-31 (M=r+8)
#pragma unroll
            for (int r = 0; r < 8; ++r)
                myLds[(r + hi * 8) * HID + j] = hreg[r];
        }
        __builtin_amdgcn_wave_barrier();

        // ---- Re-layout h row for next A operand; reuse it for the seq output store.
        v8h hrow;
        if (lane < 16) {
            hrow = *(const v8h*)(myLds + nloc * HID);
        } else {
#pragma unroll
            for (int e = 0; e < 8; ++e) hrow[e] = (_Float16)0.0f;
        }
#pragma unroll
        for (int e = 0; e < 8; ++e) hA[e] = hrow[e];   // elems 8..15 stay zero

        if constexpr (WRITE_SEQ) {
            if (lane < 16)
                *(v8h*)(out_seq + ((size_t)t * BATCH + b0 + nloc) * 16 + dirOff) = hrow;
        } else {
            if (t == TSEQ - 1 && lane < 16) {          // fwd: final step; bwd: first processed step
#pragma unroll
                for (int e = 0; e < 8; ++e)
                    out_last[(size_t)(b0 + nloc) * 64 + dirOff + e] = (float)hrow[e];
            }
        }
    }
}

// MLP head: 64 -> 32 (LN,relu) -> 16 (LN,relu) -> 2. One thread per batch row;
// weights staged in LDS. Also emits the h [B,64] output copy.
__global__ __launch_bounds__(256)
void mlp_head_kernel(const float* __restrict__ hcat,
                     const float* __restrict__ w1, const float* __restrict__ b1,
                     const float* __restrict__ g1, const float* __restrict__ bb1,
                     const float* __restrict__ w2, const float* __restrict__ b2,
                     const float* __restrict__ g2, const float* __restrict__ bb2,
                     const float* __restrict__ w3, const float* __restrict__ b3,
                     float* __restrict__ out)
{
    __shared__ float sw1[32 * 64];
    __shared__ float sw2[16 * 32];
    __shared__ float sw3[2 * 16];
    __shared__ float sb1[32], sg1[32], sbb1[32];
    __shared__ float sb2[16], sg2[16], sbb2[16];
    __shared__ float sb3[2];

    for (int i = threadIdx.x; i < 32 * 64; i += 256) sw1[i] = w1[i];
    for (int i = threadIdx.x; i < 16 * 32; i += 256) sw2[i] = w2[i];
    if (threadIdx.x < 32) {
        sw3[threadIdx.x] = w3[threadIdx.x];
        sb1[threadIdx.x] = b1[threadIdx.x];
        sg1[threadIdx.x] = g1[threadIdx.x];
        sbb1[threadIdx.x] = bb1[threadIdx.x];
    }
    if (threadIdx.x < 16) {
        sb2[threadIdx.x] = b2[threadIdx.x];
        sg2[threadIdx.x] = g2[threadIdx.x];
        sbb2[threadIdx.x] = bb2[threadIdx.x];
    }
    if (threadIdx.x < 2) sb3[threadIdx.x] = b3[threadIdx.x];
    __syncthreads();

    const int b = blockIdx.x * 256 + threadIdx.x;

    float h[64];
#pragma unroll
    for (int i = 0; i < 64; ++i) h[i] = hcat[(size_t)b * 64 + i];
#pragma unroll
    for (int i = 0; i < 64; ++i) out[(size_t)BATCH * 2 + (size_t)b * 64 + i] = h[i];

    float z1[32];
#pragma unroll
    for (int k = 0; k < 32; ++k) {
        float a = sb1[k];
        for (int i = 0; i < 64; ++i) a += sw1[k * 64 + i] * h[i];
        z1[k] = a;
    }
    float mu = 0.0f;
#pragma unroll
    for (int k = 0; k < 32; ++k) mu += z1[k];
    mu *= (1.0f / 32.0f);
    float var = 0.0f;
#pragma unroll
    for (int k = 0; k < 32; ++k) { const float d = z1[k] - mu; var += d * d; }
    var *= (1.0f / 32.0f);
    float inv = rsqrtf(var + 1e-6f);
#pragma unroll
    for (int k = 0; k < 32; ++k) {
        const float v = (z1[k] - mu) * inv * sg1[k] + sbb1[k];
        z1[k] = v > 0.0f ? v : 0.0f;
    }

    float z2[16];
#pragma unroll
    for (int k = 0; k < 16; ++k) {
        float a = sb2[k];
        for (int i = 0; i < 32; ++i) a += sw2[k * 32 + i] * z1[i];
        z2[k] = a;
    }
    mu = 0.0f;
#pragma unroll
    for (int k = 0; k < 16; ++k) mu += z2[k];
    mu *= (1.0f / 16.0f);
    var = 0.0f;
#pragma unroll
    for (int k = 0; k < 16; ++k) { const float d = z2[k] - mu; var += d * d; }
    var *= (1.0f / 16.0f);
    inv = rsqrtf(var + 1e-6f);
#pragma unroll
    for (int k = 0; k < 16; ++k) {
        const float v = (z2[k] - mu) * inv * sg2[k] + sbb2[k];
        z2[k] = v > 0.0f ? v : 0.0f;
    }

    float l0 = sb3[0], l1 = sb3[1];
#pragma unroll
    for (int i = 0; i < 16; ++i) { l0 += sw3[i] * z2[i]; l1 += sw3[16 + i] * z2[i]; }
    out[(size_t)b * 2 + 0] = l0;
    out[(size_t)b * 2 + 1] = l1;
}

extern "C" void kernel_launch(void* const* d_in, const int* in_sizes, int n_in,
                              void* d_out, int out_size, void* d_ws, size_t ws_size,
                              hipStream_t stream) {
    (void)in_sizes; (void)n_in; (void)out_size; (void)ws_size;
    const float* x = (const float*)d_in[0];

    // d_in[0]=x; per branch 3 layers x {wih_f,whh_f,bih_f,bhh_f,wih_b,whh_b,bih_b,bhh_b};
    // branches au/gaze/expr/va (bases 1,25,49,73); then mlp params at 97..106.
    _Float16* bufA = (_Float16*)d_ws;                               // [T,B,16] f16
    _Float16* bufB = bufA + (size_t)TSEQ * BATCH * 16;              // [T,B,16] f16
    float*    hcat = (float*)(bufB + (size_t)TSEQ * BATCH * 16);    // [B,64] f32

    const dim3 grid(BATCH / 16 / WPB, 2), block(WPB * 32);

#define WARGS(base, l) \
    (const float*)d_in[(base) + (l)*8 + 0], (const float*)d_in[(base) + (l)*8 + 1], \
    (const float*)d_in[(base) + (l)*8 + 2], (const float*)d_in[(base) + (l)*8 + 3], \
    (const float*)d_in[(base) + (l)*8 + 4], (const float*)d_in[(base) + (l)*8 + 5], \
    (const float*)d_in[(base) + (l)*8 + 6], (const float*)d_in[(base) + (l)*8 + 7]

#define BRANCH(D0, base, choff, hoff)                                                   \
    lstm_layer_kernel<D0, true,  true ><<<grid, block, 0, stream>>>(                    \
        x, choff, nullptr, WARGS(base, 0), bufA, nullptr);                              \
    lstm_layer_kernel<16, false, true ><<<grid, block, 0, stream>>>(                    \
        nullptr, 0, bufA, WARGS(base, 1), bufB, nullptr);                               \
    lstm_layer_kernel<16, false, false><<<grid, block, 0, stream>>>(                    \
        nullptr, 0, bufB, WARGS(base, 2), nullptr, hcat + (hoff));

    BRANCH(35, 1,  0,  0)    // au
    BRANCH(8,  25, 35, 16)   // gaze
    BRANCH(5,  49, 43, 32)   // expr
    BRANCH(2,  73, 48, 48)   // va
#undef BRANCH
#undef WARGS

    mlp_head_kernel<<<dim3(BATCH / 256), dim3(256), 0, stream>>>(
        hcat,
        (const float*)d_in[97],  (const float*)d_in[98],  (const float*)d_in[99],  (const float*)d_in[100],
        (const float*)d_in[101], (const float*)d_in[102], (const float*)d_in[103], (const float*)d_in[104],
        (const float*)d_in[105], (const float*)d_in[106],
        (float*)d_out);
}
